// FlowSAN_10582799417883
// MI455X (gfx1250) — matrix operand order repaired and checked by reading, not compiled, not run on hardware
//
#include <hip/hip_runtime.h>
#include <hip/hip_bf16.h>

// ---------------------------------------------------------------------------
// FlowSAN on gfx1250: WMMA bf16 GEMMs (LDS-staged W^T, vectorized A loads,
// fused attention-vector dots) + wave-per-node sparse softmax aggregation.
// ---------------------------------------------------------------------------

typedef __attribute__((ext_vector_type(16))) __bf16 v16bf;
typedef __attribute__((ext_vector_type(8)))  float  v8f;

#define HSTRIDE 32   // fixed row stride for all intermediate feature tables
#define DEGREE  16
#define NGRAPHS 128
#define FOUT    10

// ---------------------------------------------------------------------------
// H[N x DO] = X[N x K] @ W[K x DO]   (bf16 in, f32 accumulate, WMMA)
// One wave per 16-row tile. Compile-time K in {32,64}, DO in {10,32}.
// X row stride == K (exact). N assumed multiple of 16 (200000 is).
// If HAS_ATT: also writes s1 = H.a1, s2 = H.a2 straight from the D fragment.
// ---------------------------------------------------------------------------
template <int K, int DO, bool HAS_ATT>
__global__ __launch_bounds__(256)
void gemm_xw_wmma(const float* __restrict__ X, const float* __restrict__ W,
                  const float* __restrict__ a1, const float* __restrict__ a2,
                  float* __restrict__ H, float* __restrict__ s1,
                  float* __restrict__ s2, int N) {
  constexpr int KT = K / 32;
  constexpr int NT = (DO + 15) / 16;

  // ---- Stage W^T into LDS as bf16, zero-padded to 32 columns --------------
  __shared__ __align__(32) __bf16 WT[32 * K];   // WT[c][k]
  for (int idx = threadIdx.x; idx < 32 * K; idx += blockDim.x) {
    const int c = idx / K, k = idx % K;
    WT[idx] = (__bf16)((c < DO) ? W[k * DO + c] : 0.f);
  }
  __syncthreads();

  const int wave = (blockIdx.x * blockDim.x + threadIdx.x) >> 5;
  const int lane = threadIdx.x & 31;
  const int row0 = wave * 16;
  if (row0 >= N) return;
  const int hi = lane >> 4;      // half-wave select
  const int lr = lane & 15;

  v8f acc[NT];
#pragma unroll
  for (int nt = 0; nt < NT; ++nt)
#pragma unroll
    for (int i = 0; i < 8; ++i) acc[nt][i] = 0.f;

  // A base: lane row = lr; K window origin 0 (lanes 0-15) or 16 (lanes 16-31)
  const float* xrow = X + (size_t)(row0 + lr) * K + (hi ? 16 : 0);

#pragma unroll
  for (int kt = 0; kt < KT; ++kt) {
    // ---- A fragment: four unconditional b128 loads, pack to bf16 ----------
    const float4 x0 = *(const float4*)(xrow + kt * 32 + 0);
    const float4 x1 = *(const float4*)(xrow + kt * 32 + 4);
    const float4 x2 = *(const float4*)(xrow + kt * 32 + 8);
    const float4 x3 = *(const float4*)(xrow + kt * 32 + 12);
    v16bf af;
    af[0]  = (__bf16)x0.x; af[1]  = (__bf16)x0.y;
    af[2]  = (__bf16)x0.z; af[3]  = (__bf16)x0.w;
    af[4]  = (__bf16)x1.x; af[5]  = (__bf16)x1.y;
    af[6]  = (__bf16)x1.z; af[7]  = (__bf16)x1.w;
    af[8]  = (__bf16)x2.x; af[9]  = (__bf16)x2.y;
    af[10] = (__bf16)x2.z; af[11] = (__bf16)x2.w;
    af[12] = (__bf16)x3.x; af[13] = (__bf16)x3.y;
    af[14] = (__bf16)x3.z; af[15] = (__bf16)x3.w;

    // ---- B fragments: contiguous 32B LDS reads (K window per half) --------
    const int kOff = kt * 32 + (hi ? 16 : 0);
#pragma unroll
    for (int nt = 0; nt < NT; ++nt) {
      const int col = nt * 16 + lr;
      const v16bf bf = *(const v16bf*)(&WT[col * K + kOff]);
      acc[nt] = __builtin_amdgcn_wmma_f32_16x16x32_bf16(
          false, af, false, bf, (short)0, acc[nt], false, false);
    }
  }

  // ---- Store D: lane col = nt*16+lr; VGPR v -> row = row0 + v + 8*hi ------
#pragma unroll
  for (int nt = 0; nt < NT; ++nt) {
    const int col = nt * 16 + lr;
    if (col < DO) {
#pragma unroll
      for (int v = 0; v < 8; ++v)
        H[(size_t)(row0 + v + hi * 8) * HSTRIDE + col] = acc[nt][v];
    }
  }

  // ---- Fused attention dots: s1[r]=h_r.a1, s2[r]=h_r.a2 -------------------
  if (HAS_ATT) {
    float a1l[NT], a2l[NT];
#pragma unroll
    for (int nt = 0; nt < NT; ++nt) {
      const int col = nt * 16 + lr;
      a1l[nt] = (col < DO) ? a1[col] : 0.f;
      a2l[nt] = (col < DO) ? a2[col] : 0.f;
    }
#pragma unroll
    for (int v = 0; v < 8; ++v) {
      float t1 = 0.f, t2 = 0.f;
#pragma unroll
      for (int nt = 0; nt < NT; ++nt) {
        t1 += acc[nt][v] * a1l[nt];
        t2 += acc[nt][v] * a2l[nt];
      }
#pragma unroll
      for (int m = 8; m >= 1; m >>= 1) {   // reduce 16 cols within half-wave
        t1 += __shfl_xor(t1, m, 32);
        t2 += __shfl_xor(t2, m, 32);
      }
      if (lr == 0) {
        const int r = row0 + v + hi * 8;
        s1[r] = t1;
        s2[r] = t2;
      }
    }
  }
}

// ---------------------------------------------------------------------------
// One GAT aggregation for one node inside one wave.
// lanes 0-15 (mirrored 16-31): per-edge logits + 16-wide softmax via shfl.
// Then lane == feature: accumulate att_e * H[col_e][lane] (128B L2 rows).
// ---------------------------------------------------------------------------
__device__ __forceinline__
float gat_node_accum(int node, int lane, int DO,
                     const float* __restrict__ H,
                     const float* __restrict__ s1, const float* __restrict__ s2,
                     const int* __restrict__ col) {
  const int e = lane & 15;
  const int j = col[(size_t)node * DEGREE + e];
  const float x = s1[node] + s2[j];
  const float logit = (x > 0.f) ? x : 0.01f * x;   // leaky_relu(0.01)
  float mx = logit;
#pragma unroll
  for (int m = 8; m >= 1; m >>= 1) mx = fmaxf(mx, __shfl_xor(mx, m, 32));
  const float ex = __expf(logit - mx);
  float sum = ex;
#pragma unroll
  for (int m = 8; m >= 1; m >>= 1) sum += __shfl_xor(sum, m, 32);
  const float att = ex / sum;

  float acc = 0.f;
#pragma unroll
  for (int ee = 0; ee < DEGREE; ++ee) {
    const float w  = __shfl(att, ee, 32);
    const int   jj = __shfl(j,   ee, 32);
    const float hv = (lane < DO) ? H[(size_t)jj * HSTRIDE + lane] : 0.f;
    acc += w * hv;
  }
  return acc;
}

// ---------------------------------------------------------------------------
// h_out = relu( GATu + GATd + P ), fused; wave per node; in-place safe.
// ---------------------------------------------------------------------------
__global__ __launch_bounds__(256)
void aggregate_combine(const float* __restrict__ Hu, const float* __restrict__ s1u,
                       const float* __restrict__ s2u, const int* __restrict__ lu_col,
                       const float* __restrict__ Hd, const float* __restrict__ s1d,
                       const float* __restrict__ s2d, const int* __restrict__ ld_col,
                       const float* __restrict__ Hp, const int* __restrict__ p_col,
                       const float* __restrict__ p_val,
                       float* __restrict__ Hout, int N, int DO) {
  const int node = (blockIdx.x * blockDim.x + threadIdx.x) >> 5;
  const int lane = threadIdx.x & 31;
  if (node >= N) return;

  float acc = gat_node_accum(node, lane, DO, Hu, s1u, s2u, lu_col)
            + gat_node_accum(node, lane, DO, Hd, s1d, s2d, ld_col);

  // p-layer: segment_sum(p_val * Hp[p_col])
  const int e  = lane & 15;
  const int jp = p_col[(size_t)node * DEGREE + e];
  const float pv = p_val[(size_t)node * DEGREE + e];
#pragma unroll
  for (int ee = 0; ee < DEGREE; ++ee) {
    const float w  = __shfl(pv, ee, 32);
    const int   jj = __shfl(jp, ee, 32);
    const float hv = (lane < DO) ? Hp[(size_t)jj * HSTRIDE + lane] : 0.f;
    acc += w * hv;
  }
  const float r = (acc > 0.f) ? acc : 0.f;
  Hout[(size_t)node * HSTRIDE + lane] = (lane < DO) ? r : 0.f;
}

// ---------------------------------------------------------------------------
// Pool |h| by graph: LDS-staged atomics, few global atomics per block.
// ---------------------------------------------------------------------------
__global__ __launch_bounds__(256)
void pool_abs(const float* __restrict__ H, const int* __restrict__ batch,
              float* __restrict__ sums, float* __restrict__ cnts, int N) {
  __shared__ float ls[NGRAPHS * FOUT];
  __shared__ float lc[NGRAPHS];
  for (int i = threadIdx.x; i < NGRAPHS * FOUT; i += blockDim.x) ls[i] = 0.f;
  for (int i = threadIdx.x; i < NGRAPHS; i += blockDim.x) lc[i] = 0.f;
  __syncthreads();
  for (int i = blockIdx.x * blockDim.x + threadIdx.x; i < N;
       i += gridDim.x * blockDim.x) {
    const int b = batch[i];
#pragma unroll
    for (int f = 0; f < FOUT; ++f)
      atomicAdd(&ls[b * FOUT + f], fabsf(H[(size_t)i * HSTRIDE + f]));
    atomicAdd(&lc[b], 1.f);
  }
  __syncthreads();
  for (int i = threadIdx.x; i < NGRAPHS * FOUT; i += blockDim.x)
    if (ls[i] != 0.f) atomicAdd(&sums[i], ls[i]);
  for (int i = threadIdx.x; i < NGRAPHS; i += blockDim.x)
    if (lc[i] != 0.f) atomicAdd(&cnts[i], lc[i]);
}

__global__ void finalize_softmax(const float* __restrict__ sums,
                                 const float* __restrict__ cnts,
                                 float* __restrict__ out) {
  const int g = threadIdx.x;
  if (g >= NGRAPHS) return;
  const float c = fmaxf(cnts[g], 1.f);
  float v[FOUT];
  float mx = -3.4e38f;
#pragma unroll
  for (int f = 0; f < FOUT; ++f) { v[f] = sums[g * FOUT + f] / c; mx = fmaxf(mx, v[f]); }
  float s = 0.f;
#pragma unroll
  for (int f = 0; f < FOUT; ++f) { v[f] = __expf(v[f] - mx); s += v[f]; }
#pragma unroll
  for (int f = 0; f < FOUT; ++f) out[g * FOUT + f] = v[f] / s;
}

// ---------------------------------------------------------------------------
extern "C" void kernel_launch(void* const* d_in, const int* in_sizes, int n_in,
                              void* d_out, int out_size, void* d_ws, size_t ws_size,
                              hipStream_t stream) {
  const float* x      = (const float*)d_in[0];
  const int*   lu_col = (const int*)  d_in[2];
  const int*   ld_col = (const int*)  d_in[4];
  const int*   p_col  = (const int*)  d_in[6];
  const float* p_val  = (const float*)d_in[7];
  const int*   batch  = (const int*)  d_in[8];

  const int N = in_sizes[8];              // 200000

  // workspace layout (floats)
  float* ws = (float*)d_ws;
  size_t off = 0;
  float* Hu   = ws + off; off += (size_t)N * HSTRIDE;
  float* Hd   = ws + off; off += (size_t)N * HSTRIDE;
  float* Hp   = ws + off; off += (size_t)N * HSTRIDE;
  float* Hcur = ws + off; off += (size_t)N * HSTRIDE;
  float* s1u  = ws + off; off += N;
  float* s2u  = ws + off; off += N;
  float* s1d  = ws + off; off += N;
  float* s2d  = ws + off; off += N;
  float* sums = ws + off; off += NGRAPHS * FOUT;
  float* cnts = ws + off; off += NGRAPHS;

  hipMemsetAsync(sums, 0, (NGRAPHS * FOUT + NGRAPHS) * sizeof(float), stream);

  const int gemmWaves  = (N + 15) / 16;
  const int gemmBlocks = (gemmWaves * 32 + 255) / 256;
  const int nodeBlocks = (int)(((size_t)N * 32 + 255) / 256);

  for (int L = 0; L < 4; ++L) {
    const float* hin = (L == 0) ? x : Hcur;
    const int base = 9 + L * 7;  // params flattened: Wu,au1,au2,Wd,ad1,ad2,Wp
    const float* Wu  = (const float*)d_in[base + 0];
    const float* au1 = (const float*)d_in[base + 1];
    const float* au2 = (const float*)d_in[base + 2];
    const float* Wd  = (const float*)d_in[base + 3];
    const float* ad1 = (const float*)d_in[base + 4];
    const float* ad2 = (const float*)d_in[base + 5];
    const float* Wp  = (const float*)d_in[base + 6];

    if (L == 0) {
      gemm_xw_wmma<64, 32, true ><<<gemmBlocks, 256, 0, stream>>>(hin, Wu, au1, au2, Hu, s1u, s2u, N);
      gemm_xw_wmma<64, 32, true ><<<gemmBlocks, 256, 0, stream>>>(hin, Wd, ad1, ad2, Hd, s1d, s2d, N);
      gemm_xw_wmma<64, 32, false><<<gemmBlocks, 256, 0, stream>>>(hin, Wp, nullptr, nullptr, Hp, nullptr, nullptr, N);
    } else if (L < 3) {
      gemm_xw_wmma<32, 32, true ><<<gemmBlocks, 256, 0, stream>>>(hin, Wu, au1, au2, Hu, s1u, s2u, N);
      gemm_xw_wmma<32, 32, true ><<<gemmBlocks, 256, 0, stream>>>(hin, Wd, ad1, ad2, Hd, s1d, s2d, N);
      gemm_xw_wmma<32, 32, false><<<gemmBlocks, 256, 0, stream>>>(hin, Wp, nullptr, nullptr, Hp, nullptr, nullptr, N);
    } else {
      gemm_xw_wmma<32, FOUT, true ><<<gemmBlocks, 256, 0, stream>>>(hin, Wu, au1, au2, Hu, s1u, s2u, N);
      gemm_xw_wmma<32, FOUT, true ><<<gemmBlocks, 256, 0, stream>>>(hin, Wd, ad1, ad2, Hd, s1d, s2d, N);
      gemm_xw_wmma<32, FOUT, false><<<gemmBlocks, 256, 0, stream>>>(hin, Wp, nullptr, nullptr, Hp, nullptr, nullptr, N);
    }

    const int DO = (L == 3) ? FOUT : 32;
    aggregate_combine<<<nodeBlocks, 256, 0, stream>>>(
        Hu, s1u, s2u, lu_col, Hd, s1d, s2d, ld_col,
        Hp, p_col, p_val, Hcur, N, DO);
  }

  pool_abs<<<512, 256, 0, stream>>>(Hcur, batch, sums, cnts, N);
  finalize_softmax<<<1, NGRAPHS, 0, stream>>>(sums, cnts, (float*)d_out);
}
